// ResBlockBitNet_17978733101750
// MI455X (gfx1250) — compile-verified, matrix operand order checked by reading, not run on hardware
//
#include <hip/hip_runtime.h>
#include <hip/hip_bf16.h>
#include <math.h>

// Problem dims
#define C_DIM 512
#define L_DIM 4096
#define B_DIM 16
#define LP    (L_DIM + 2)   // padded length (zero row at 0 and LP-1)

typedef __attribute__((ext_vector_type(16))) _Float16 v16h;
typedef __attribute__((ext_vector_type(8)))  _Float16 v8h;
typedef __attribute__((ext_vector_type(8)))  float    v8f;

typedef unsigned int u32x4 __attribute__((ext_vector_type(4)));
typedef int          i32x8 __attribute__((ext_vector_type(8)));
typedef int          i32x4 __attribute__((ext_vector_type(4)));

union Frag16 { v16h v; struct { v8h lo, hi; } p; };

__device__ __forceinline__ float silu_f(float v) {
    return v / (1.f + __expf(-v));
}

// ---------------------------------------------------------------------------
// Zero the conv boundary pad rows of both fp16 activation buffers.
// ---------------------------------------------------------------------------
__global__ void zero_pads_kernel(_Float16* __restrict__ hpad,
                                 _Float16* __restrict__ h2pad) {
    int idx = blockIdx.x * 256 + threadIdx.x;          // 16*2*512 = 16384 total
    if (idx >= B_DIM * 2 * C_DIM) return;
    int b   = idx / (2 * C_DIM);
    int rc  = idx % (2 * C_DIM);
    int row = (rc < C_DIM) ? 0 : (LP - 1);
    int c   = rc & (C_DIM - 1);
    size_t off = ((size_t)b * LP + row) * C_DIM + c;
    hpad[off]  = (_Float16)0.f;
    h2pad[off] = (_Float16)0.f;
}

// ---------------------------------------------------------------------------
// Stage 1 of deterministic mean(|w|): 768 blocks x 1024 elems per weight.
// ---------------------------------------------------------------------------
__global__ void wsum_abs_kernel(const float* __restrict__ w1,
                                const float* __restrict__ w2,
                                float* __restrict__ partials) {
    const float* w = (blockIdx.y == 0) ? w1 : w2;
    __shared__ float red[256];
    int base = blockIdx.x * 1024;                      // 768*1024 = 786432 exact
    float s = 0.f;
    for (int j = threadIdx.x; j < 1024; j += 256)
        s += fabsf(w[base + j]);
    red[threadIdx.x] = s;
    __syncthreads();
    for (int off = 128; off > 0; off >>= 1) {
        if (threadIdx.x < off) red[threadIdx.x] += red[threadIdx.x + off];
        __syncthreads();
    }
    if (threadIdx.x == 0) partials[blockIdx.y * 768 + blockIdx.x] = red[0];
}

// Stage 2: reduce 768 partials -> scale per weight.
__global__ void wscale_kernel(const float* __restrict__ partials,
                              float* __restrict__ scales) {
    __shared__ float red[256];
    float s = 0.f;
    for (int j = threadIdx.x; j < 768; j += 256)
        s += partials[blockIdx.x * 768 + j];
    red[threadIdx.x] = s;
    __syncthreads();
    for (int off = 128; off > 0; off >>= 1) {
        if (threadIdx.x < off) red[threadIdx.x] += red[threadIdx.x + off];
        __syncthreads();
    }
    if (threadIdx.x == 0)
        scales[blockIdx.x] = fmaxf(red[0] * (1.f / 786432.f), 1e-5f);
}

// ---------------------------------------------------------------------------
// Ternary quantize w[O][I][K] (f32) -> wq[K][O][I] (fp16), A-operand layout.
// ---------------------------------------------------------------------------
__global__ void wquant_kernel(const float* __restrict__ w1,
                              const float* __restrict__ w2,
                              const float* __restrict__ scales,
                              _Float16* __restrict__ wq1,
                              _Float16* __restrict__ wq2) {
    const float* w  = (blockIdx.y == 0) ? w1 : w2;
    _Float16*    wq = (blockIdx.y == 0) ? wq1 : wq2;
    float s   = scales[blockIdx.y];
    float inv = 1.f / s;
    int idx = blockIdx.x * 256 + threadIdx.x;          // 0 .. 262143 = (o,i)
    int o = idx >> 9;
    int i = idx & 511;
    #pragma unroll
    for (int k = 0; k < 3; ++k) {
        float wv = w[(size_t)idx * 3 + k];
        float q  = fminf(fmaxf(rintf(wv * inv), -1.f), 1.f) * s;
        wq[((size_t)k * C_DIM + o) * C_DIM + i] = (_Float16)q;
    }
}

// ---------------------------------------------------------------------------
// RMSNorm(channel) + SiLU + transpose: x[B][C][L] f32 -> hpad[B][LP][C] fp16.
// ---------------------------------------------------------------------------
__global__ void norm_silu_kernel(const float* __restrict__ x,
                                 const float* __restrict__ g,
                                 _Float16* __restrict__ hpad) {
    __shared__ float red[256];
    const int lt = threadIdx.x & 63;       // l within block
    const int q  = threadIdx.x >> 6;       // channel quarter (128 ch each)
    const int b  = blockIdx.y;
    const int l  = blockIdx.x * 64 + lt;
    const float* xb = x + (size_t)b * C_DIM * L_DIM + l;

    float ss = 0.f;
    for (int c = q * 128; c < q * 128 + 128; ++c) {
        float v = xb[(size_t)c * L_DIM];
        ss += v * v;
    }
    red[threadIdx.x] = ss;
    __syncthreads();
    float tot = red[lt] + red[64 + lt] + red[128 + lt] + red[192 + lt];
    float r = rsqrtf(tot * (1.f / (float)C_DIM) + 1e-6f);

    _Float16* hrow = hpad + ((size_t)b * LP + l + 1) * C_DIM;
    for (int c0 = q * 128; c0 < q * 128 + 128; c0 += 8) {
        v8h hv;
        #pragma unroll
        for (int j = 0; j < 8; ++j) {
            float v = xb[(size_t)(c0 + j) * L_DIM] * r * g[c0 + j];
            hv[j] = (_Float16)silu_f(v);
        }
        *(v8h*)(hrow + c0) = hv;
    }
}

// ---------------------------------------------------------------------------
// TDM issue: 2D tile 512 x 66 fp16, LDS pad 4 DWORDs per 256-DWORD row
// -> LDS row stride 1040B (bank-skewed, 16B aligned).  (ISA 8.3-8.5)
// ---------------------------------------------------------------------------
#define TILE_ROWS 66
#define TILE_RSTR 520   // halves; 1040B rows = 512 elems + 16B TDM pad
#define NITER     2     // l-chunks per block (double buffered)

__device__ __forceinline__ void tdm_issue_tile(const _Float16* src,
                                               unsigned int lds_base) {
    unsigned long long ga = (unsigned long long)src;
    u32x4 g0;
    g0[0] = 1u;                                   // count=1 (valid, user)
    g0[1] = lds_base;                             // lds_addr (bytes)
    g0[2] = (unsigned int)(ga & 0xFFFFFFFFull);   // global_addr[31:0]
    g0[3] = (unsigned int)((ga >> 32) & 0x1FFFFFFull)
          | (2u << 30);                           // global_addr[56:32], type=2
    i32x8 g1;
    g1[0] = (1 << 16)      // data_size = 2 bytes
          | (1 << 20)      // pad_enable
          | (7 << 22)      // pad_interval = 256 DWORDs (one 1024B row)
          | (3 << 25);     // pad_amount   = 4 DWORDs (16B skew)
    g1[1] = (int)(512u << 16);   // tensor_dim0 = 512   (bits 79:48 lo)
    g1[2] = (int)(66u << 16);    // tensor_dim1 = 66    (bits 111:80 lo)
    g1[3] = (int)(512u << 16);   // tile_dim0   = 512   (bits 127:112)
    g1[4] = 66;                  // tile_dim1   = 66    (bits 143:128)
    g1[5] = 512;                 // tensor_dim0_stride = 512 (bits 207:160 lo)
    g1[6] = 0;
    g1[7] = 0;
    i32x4 g2 = {0, 0, 0, 0};
    i32x4 g3 = {0, 0, 0, 0};
    i32x8 g4 = {0, 0, 0, 0, 0, 0, 0, 0};
    __builtin_amdgcn_tensor_load_to_lds(g0, g1, g2, g3, g4, 0);
}

// ---------------------------------------------------------------------------
// WMMA conv-as-GEMM, double-buffered TDM staging.
//   Block: 512 threads = 16 waves; block tile ALL 512 o x 64 l, NITER chunks.
//   Wave:  32 o x 64 l => 8 v8f accumulators; each LDS B fragment feeds
//          two WMMAs.  wmma/wave/chunk: 3 x 16 x 8 = 384.
// STAGE 1: +bias, SiLU, write fp16 transposed padded layout (input of GEMM2).
// STAGE 2: +bias +residual, write f32 [B][C][L].
// ---------------------------------------------------------------------------
template <int STAGE>
__global__ __launch_bounds__(512)
void conv_wmma_kernel(const _Float16* __restrict__ hin,
                      const _Float16* __restrict__ wq,
                      const float* __restrict__ bias,
                      const float* __restrict__ xres,
                      _Float16* __restrict__ hout,
                      float* __restrict__ yout) {
    extern __shared__ _Float16 tile[];   // 2 * TILE_ROWS * TILE_RSTR halves

    const int tid   = threadIdx.x;
    const int lane  = tid & 31;
    const int wave  = tid >> 5;          // 0..15
    const int ln    = lane & 15;         // M row (A) / N col (B,D) index
    const int hh    = lane >> 4;         // K half-chunk select (16-bit layout)
    const int b     = blockIdx.z;
    const int lbase = blockIdx.x * (64 * NITER);
    const int o0    = wave * 32;

    _Float16* bufp[2] = { tile, tile + TILE_ROWS * TILE_RSTR };
    unsigned int lds0 = (unsigned int)(unsigned long long)
        (__attribute__((address_space(3))) _Float16*)bufp[0];
    unsigned int lds1 = (unsigned int)(unsigned long long)
        (__attribute__((address_space(3))) _Float16*)bufp[1];
    const _Float16* Hblk = hin + ((size_t)b * LP + lbase) * C_DIM;

    // prologue: start DMA of chunk 0 into buffer 0
    if (wave == 0) tdm_issue_tile(Hblk, lds0);

    #pragma unroll
    for (int it = 0; it < NITER; ++it) {
        const int l0 = lbase + it * 64;
        // start DMA of next chunk into other buffer, wait for current chunk
        if (wave == 0) {
            if (it + 1 < NITER) {
                tdm_issue_tile(Hblk + (size_t)(it + 1) * 64 * C_DIM,
                               (it & 1) ? lds0 : lds1);
                __builtin_amdgcn_s_wait_tensorcnt(1);
            } else {
                __builtin_amdgcn_s_wait_tensorcnt(0);
            }
        }
        __syncthreads();
        const _Float16* cur = bufp[it & 1];

        v8f zero = {0.f, 0.f, 0.f, 0.f, 0.f, 0.f, 0.f, 0.f};
        v8f acc[2][4];
        #pragma unroll
        for (int oi = 0; oi < 2; ++oi)
            #pragma unroll
            for (int t = 0; t < 4; ++t) acc[oi][t] = zero;

        #pragma unroll
        for (int tap = 0; tap < 3; ++tap) {
            // output l uses padded rows (l + tap); LDS row = (l - l0) + tap
            const _Float16* Ab0 =
                wq + ((size_t)tap * C_DIM + o0 + ln) * C_DIM + hh * 8;
            const _Float16* Ab1 = Ab0 + (size_t)16 * C_DIM;
            for (int i0 = 0; i0 < C_DIM; i0 += 32) {
                Frag16 A0, A1;
                A0.p.lo = *(const v8h*)(Ab0 + i0);
                A0.p.hi = *(const v8h*)(Ab0 + i0 + 16);
                A1.p.lo = *(const v8h*)(Ab1 + i0);
                A1.p.hi = *(const v8h*)(Ab1 + i0 + 16);
                __builtin_prefetch(Ab0 + i0 + 64, 0, 1);  // global_prefetch_b8
                #pragma unroll
                for (int t = 0; t < 4; ++t) {
                    const _Float16* Bp =
                        cur + (tap + t * 16 + ln) * TILE_RSTR + i0 + hh * 8;
                    Frag16 Bf;
                    Bf.p.lo = *(const v8h*)(Bp);          // ds_load_b128
                    Bf.p.hi = *(const v8h*)(Bp + 16);     // ds_load_b128
                    acc[0][t] = __builtin_amdgcn_wmma_f32_16x16x32_f16(
                        false, A0.v, false, Bf.v, (short)0, acc[0][t],
                        false, false);
                    acc[1][t] = __builtin_amdgcn_wmma_f32_16x16x32_f16(
                        false, A1.v, false, Bf.v, (short)0, acc[1][t],
                        false, false);
                }
            }
        }

        // D layout: VGPR r holds M = r + 8*hh (o), lane ln holds N (l).
        #pragma unroll
        for (int oi = 0; oi < 2; ++oi) {
            const int ob = o0 + oi * 16 + 8 * hh;
            float bv[8];
            #pragma unroll
            for (int r = 0; r < 8; ++r) bv[r] = bias[ob + r];

            if (STAGE == 1) {
                #pragma unroll
                for (int t = 0; t < 4; ++t) {
                    v8h outv;
                    #pragma unroll
                    for (int r = 0; r < 8; ++r) {
                        float v = acc[oi][t][r] + bv[r];
                        outv[r] = (_Float16)silu_f(v);
                    }
                    _Float16* dst = hout +
                        ((size_t)b * LP + (l0 + 16 * t + ln) + 1) * C_DIM + ob;
                    *(v8h*)dst = outv;
                }
            } else {
                #pragma unroll
                for (int t = 0; t < 4; ++t) {
                    const int l = l0 + 16 * t + ln;
                    #pragma unroll
                    for (int r = 0; r < 8; ++r) {
                        size_t oix = ((size_t)b * C_DIM + ob + r) * L_DIM + l;
                        yout[oix] = acc[oi][t][r] + bv[r] + xres[oix];
                    }
                }
            }
        }
        __syncthreads();   // protect buffer (it&1) before next DMA reuses it
    }
}

// ---------------------------------------------------------------------------
extern "C" void kernel_launch(void* const* d_in, const int* in_sizes, int n_in,
                              void* d_out, int out_size, void* d_ws, size_t ws_size,
                              hipStream_t stream) {
    (void)in_sizes; (void)n_in; (void)out_size; (void)ws_size;
    const float* x  = (const float*)d_in[0];
    const float* w1 = (const float*)d_in[1];
    const float* b1 = (const float*)d_in[2];
    const float* w2 = (const float*)d_in[3];
    const float* b2 = (const float*)d_in[4];
    const float* g  = (const float*)d_in[5];
    float* out = (float*)d_out;

    // Workspace carving (all 256B-aligned sections)
    char* p = (char*)d_ws;
    const size_t hpad_elems = (size_t)B_DIM * LP * C_DIM;      // 33,570,816
    _Float16* hpad  = (_Float16*)p;  p += hpad_elems * 2;      // 67.1 MB
    _Float16* h2pad = (_Float16*)p;  p += hpad_elems * 2;      // 67.1 MB
    _Float16* wq1   = (_Float16*)p;  p += (size_t)3 * C_DIM * C_DIM * 2;
    _Float16* wq2   = (_Float16*)p;  p += (size_t)3 * C_DIM * C_DIM * 2;
    float* partials = (float*)p;     p += 2 * 768 * sizeof(float);
    float* scales   = (float*)p;

    const size_t tile_bytes =
        (size_t)2 * TILE_ROWS * TILE_RSTR * sizeof(_Float16);  // 137,280 B

    zero_pads_kernel<<<dim3(64), 256, 0, stream>>>(hpad, h2pad);
    wsum_abs_kernel<<<dim3(768, 2), 256, 0, stream>>>(w1, w2, partials);
    wscale_kernel<<<dim3(2), 256, 0, stream>>>(partials, scales);
    wquant_kernel<<<dim3(1024, 2), 256, 0, stream>>>(w1, w2, scales, wq1, wq2);
    norm_silu_kernel<<<dim3(64, 16), 256, 0, stream>>>(x, g, hpad);
    conv_wmma_kernel<1><<<dim3(32, 1, 16), 512, tile_bytes, stream>>>(
        hpad, wq1, b1, nullptr, h2pad, nullptr);
    conv_wmma_kernel<2><<<dim3(32, 1, 16), 512, tile_bytes, stream>>>(
        h2pad, wq2, b2, x, nullptr, out);
}
